// AllAtomEquivariantAtomAttention_78142634983965
// MI455X (gfx1250) — matrix operand order repaired and checked by reading, not compiled, not run on hardware
//
#include <hip/hip_runtime.h>
#include <hip/hip_bf16.h>
#include <math.h>

// ---------------- problem constants (fixed by reference) ----------------
#define E_NUM   8192
#define FLATN   256
#define NE      64
#define ATOM    128
#define EDIM    32
#define ZEMB    32
#define RBFN    16
#define HIDN    256
#define LATN    128
#define MUL0    32
#define MUL1    16
#define OUTD    80
#define INVD    48
#define WNUM    2304
#define CUT     5.0f
// padded K dims (multiples of 32 for WMMA K-steps)
#define WIN_K   64    // weight_in 49 -> 64
#define PS_K    192   // pair_static 177 -> 192
#define QIN_K   160   // q input 160
#define INV_K   64    // inv 48 -> 64

typedef __attribute__((ext_vector_type(16))) _Float16 v16h;
typedef __attribute__((ext_vector_type(8)))  _Float16 v8h;
typedef __attribute__((ext_vector_type(8)))  float    v8f;

// ------------- ordered-float encode for atomicMax on u32 -------------
__device__ __forceinline__ unsigned int fenc_d(float f) {
  unsigned int u = __float_as_uint(f);
  return (u & 0x80000000u) ? ~u : (u | 0x80000000u);
}
__device__ __forceinline__ float fdec_d(unsigned int u) {
  u = (u & 0x80000000u) ? (u & 0x7fffffffu) : ~u;
  return __uint_as_float(u);
}
static unsigned int fenc_h(float f) {
  unsigned int u; __builtin_memcpy(&u, &f, 4);
  return (u & 0x80000000u) ? ~u : (u | 0x80000000u);
}

// ------------- generic fill / convert -------------
__global__ void fill_u32(unsigned int* p, unsigned int v, int n) {
  int i = blockIdx.x * blockDim.x + threadIdx.x;
  if (i < n) p[i] = v;
}
__global__ void cvt_f16(const float* s, _Float16* d, int n) {
  int i = blockIdx.x * blockDim.x + threadIdx.x;
  if (i < n) d[i] = (_Float16)s[i];
}

// ------------- pack f32 weight (Korig x N, row-major) into WMMA B-fragment layout -------------
// layout: [(tileN*(Kpad/32)+ks)*512 + lane*16 + j], K=ks*32+(j>=8?16:0)+(lane>=16?8:0)+(j&7) ; N=tileN*16+lane%16
__global__ void pack_w(const float* W, _Float16* Wp, int Korig, int Kpad, int N) {
  int idx = blockIdx.x * blockDim.x + threadIdx.x;
  int per_tile = (Kpad >> 5) * 512;
  int total = (N >> 4) * per_tile;
  if (idx >= total) return;
  int tn = idx / per_tile;
  int r = idx - tn * per_tile;
  int ks = r >> 9;
  int r2 = r & 511;
  int lane = r2 >> 4, j = r2 & 15;
  int n = (tn << 4) + (lane & 15);
  int k = (ks << 5) + ((j & 8) << 1) + ((lane >> 4) << 3) + (j & 7);
  float v = (k < Korig) ? W[(size_t)k * N + n] : 0.0f;
  Wp[idx] = (_Float16)v;
}

// build v16h A fragment per ISA layout from two aligned 16B chunks
__device__ __forceinline__ v16h load_a_frag(const _Float16* Arow, int ks) {
  v8h a0 = *(const v8h*)(Arow + (ks << 5));        // K = k0+off .. +7   -> elements 0..7
  v8h a1 = *(const v8h*)(Arow + (ks << 5) + 16);   // K = k0+off+16..+23 -> elements 8..15
  v16h a;
#pragma unroll
  for (int j = 0; j < 8; ++j) { a[j] = a0[j]; a[8 + j] = a1[j]; }
  return a;
}

__device__ __forceinline__ void store_cd(void* Cout, const v8f& acc, const float* bias,
                                         int tm16, int n, int laneHi, int N,
                                         bool silu, bool outf16) {
  float bn = bias ? bias[n] : 0.0f;
#pragma unroll
  for (int r = 0; r < 8; ++r) {
    int mr = tm16 + r + (laneHi ? 8 : 0);
    float v = acc[r] + bn;
    if (silu) v = v / (1.0f + __expf(-v));
    if (outf16) ((_Float16*)Cout)[(size_t)mr * N + n] = (_Float16)v;
    else        ((float*)Cout)[(size_t)mr * N + n] = v;
  }
}

// ------------- simple WMMA GEMM (fallback for small/odd shapes): one wave per 16x16 tile -------------
template <bool SILU, bool OUTF16>
__global__ void gemm_wmma(const _Float16* __restrict__ A, const _Float16* __restrict__ Wp,
                          const float* __restrict__ bias, void* __restrict__ Cout,
                          int M, int Kpad, int N) {
  int wave = threadIdx.x >> 5;
  int lane = threadIdx.x & 31;
  int tiles_n = N >> 4;
  int tile = blockIdx.x * (blockDim.x >> 5) + wave;
  if (tile >= (M >> 4) * tiles_n) return;          // wave-uniform: EXEC stays all-ones
  int tm = tile / tiles_n, tn = tile - tm * tiles_n;
  int laneHi = lane >> 4;
  int m = (tm << 4) + (lane & 15);
  v8f acc = {};
  const _Float16* Arow = A + (size_t)m * Kpad + (laneHi ? 8 : 0);
  const _Float16* Bp = Wp + ((size_t)tn * (Kpad >> 5)) * 512 + lane * 16;
  int nks = Kpad >> 5;
  for (int ks = 0; ks < nks; ++ks) {
    v16h a = load_a_frag(Arow, ks);
    v16h b = *(const v16h*)(Bp + (size_t)ks * 512);
    acc = __builtin_amdgcn_wmma_f32_16x16x32_f16(false, a, false, b, (short)0, acc, false, false);
  }
  store_cd(Cout, acc, bias, tm << 4, (tn << 4) + (lane & 15), laneHi, N, SILU, OUTF16);
}

// ------------- macro-tiled WMMA GEMM: 128x64 per block, B panel staged in LDS -------------
// 8 waves: wave w -> rows [bm*128 + w*16, +16), cols [bn*64, +64) as 4 accumulators.
// B panel (4 N-tiles x Kpad, <=32KB) staged once per block, read via ds_load_b128.
template <bool SILU, bool OUTF16>
__global__ void gemm_wmma_blk(const _Float16* __restrict__ A, const _Float16* __restrict__ Wp,
                              const float* __restrict__ bias, void* __restrict__ Cout,
                              int M, int Kpad, int N) {
  __shared__ _Float16 sB[8 * 2048];                // K up to 256: 8 ksteps * (4 tiles * 512 halves)
  int nks = Kpad >> 5;
  int nblk = N >> 6;
  int bm = blockIdx.x / nblk, bn = blockIdx.x - bm * nblk;
  // cooperative stage: nks*2048 halves in 16B chunks
  int chunks = nks << 8;
  for (int c = threadIdx.x; c < chunks; c += 256) {
    int f = c << 3;                                 // half index
    int ks = f >> 11;
    int r = f & 2047;
    int ltn = r >> 9;
    int within = r & 511;
    const v8h* src = (const v8h*)(Wp + (((size_t)(bn * 4 + ltn) * nks + ks) << 9) + within);
    *(v8h*)(sB + f) = *src;
  }
  __syncthreads();
  int wave = threadIdx.x >> 5, lane = threadIdx.x & 31;
  int laneHi = lane >> 4;
  int m = (bm << 7) + (wave << 4) + (lane & 15);
  const _Float16* Arow = A + (size_t)m * Kpad + (laneHi ? 8 : 0);
  v8f acc0 = {}, acc1 = {}, acc2 = {}, acc3 = {};
  for (int ks = 0; ks < nks; ++ks) {
    if (ks + 1 < nks) __builtin_prefetch((const void*)(Arow + ((ks + 1) << 5)), 0, 0);
    v16h a = load_a_frag(Arow, ks);
    const _Float16* bb = sB + (ks << 11) + lane * 16;
    v16h b0 = *(const v16h*)(bb);
    v16h b1 = *(const v16h*)(bb + 512);
    v16h b2 = *(const v16h*)(bb + 1024);
    v16h b3 = *(const v16h*)(bb + 1536);
    acc0 = __builtin_amdgcn_wmma_f32_16x16x32_f16(false, a, false, b0, (short)0, acc0, false, false);
    acc1 = __builtin_amdgcn_wmma_f32_16x16x32_f16(false, a, false, b1, (short)0, acc1, false, false);
    acc2 = __builtin_amdgcn_wmma_f32_16x16x32_f16(false, a, false, b2, (short)0, acc2, false, false);
    acc3 = __builtin_amdgcn_wmma_f32_16x16x32_f16(false, a, false, b3, (short)0, acc3, false, false);
  }
  int tm16 = (bm << 7) + (wave << 4);
  int nbase = (bn << 6) + (lane & 15);
  store_cd(Cout, acc0, bias, tm16, nbase,      laneHi, N, SILU, OUTF16);
  store_cd(Cout, acc1, bias, tm16, nbase + 16, laneHi, N, SILU, OUTF16);
  store_cd(Cout, acc2, bias, tm16, nbase + 32, laneHi, N, SILU, OUTF16);
  store_cd(Cout, acc3, bias, tm16, nbase + 48, laneHi, N, SILU, OUTF16);
}

// ------------- per-edge geometric features + MLP input rows -------------
__global__ void edge_prep(const float* att_dist, const float* att_vec,
                          const int* att_src, const int* att_dst,
                          const int* z, const unsigned char* mask,
                          const float* z_emb, const float* h,
                          _Float16* win, _Float16* ps,
                          float* sh1e, float* enve, float* active) {
  int e = blockIdx.x * blockDim.x + threadIdx.x;
  if (e >= E_NUM) return;
  int src = att_src[e], dst = att_dst[e];
  float d = fmaxf(att_dist[e], 1e-8f);
  bool self = (src == dst);
  float invd = 1.0f / d;
  const float SQ3 = 1.7320508075688772f;
#pragma unroll
  for (int c = 0; c < 3; ++c) {
    float u = att_vec[e * 3 + c] * invd;
    sh1e[e * 3 + c] = self ? 0.0f : SQ3 * u;
  }
  enve[e] = (d < CUT) ? 0.5f * (__cosf(3.14159265358979f * d / CUT) + 1.0f) : 0.0f;
  active[e] = (mask[src] && mask[dst]) ? 1.0f : 0.0f;
  const float* zr = z_emb + (size_t)z[dst] * ZEMB;
  float rbf[RBFN];
  const float width = CUT / (RBFN - 1);
#pragma unroll
  for (int i = 0; i < RBFN; ++i) {
    float t = d - (float)i * width;
    rbf[i] = __expf(-t * t / (2.0f * width * width));
  }
  float selfF = self ? 1.0f : 0.0f;
  _Float16* w = win + (size_t)e * WIN_K;
  for (int i = 0; i < ZEMB; ++i) w[i] = (_Float16)zr[i];
  w[ZEMB] = (_Float16)selfF;
  for (int i = 0; i < RBFN; ++i) w[ZEMB + 1 + i] = (_Float16)rbf[i];
  for (int i = ZEMB + 1 + RBFN; i < WIN_K; ++i) w[i] = (_Float16)0.0f;
  _Float16* p = ps + (size_t)e * PS_K;
  const float* hd = h + (size_t)dst * ATOM;
  for (int i = 0; i < ATOM; ++i) p[i] = (_Float16)hd[i];
  for (int i = 0; i < ZEMB; ++i) p[ATOM + i] = (_Float16)zr[i];
  p[ATOM + ZEMB] = (_Float16)selfF;
  for (int i = 0; i < RBFN; ++i) p[ATOM + ZEMB + 1 + i] = (_Float16)rbf[i];
  for (int i = ATOM + ZEMB + 1 + RBFN; i < PS_K; ++i) p[i] = (_Float16)0.0f;
}

// ------------- per-edge tensor product (small per-edge matvecs on tp_w) -------------
__global__ void tp_kernel(const int* att_dst, const float* h_full,
                          const _Float16* tpw, const float* sh1e, const float* enve,
                          float* w_edge) {
  int idx = blockIdx.x * blockDim.x + threadIdx.x;
  if (idx >= E_NUM * OUTD) return;
  int e = idx / OUTD, j = idx - e * OUTD;
  const float* hf = h_full + (size_t)att_dst[e] * OUTD;
  const _Float16* w = tpw + (size_t)e * WNUM;
  const float alpha = 0.14433756729740643f;   // 1/sqrt(48)
  const float inv_sq3 = 0.5773502691896258f;
  float out;
  if (j < MUL0) {
    int o = j;
    float acc = 0.f;
    for (int i = 0; i < MUL0; ++i) acc += hf[i] * (float)w[i * MUL0 + o];                  // w1
    float acc4 = 0.f;
    for (int i = 0; i < MUL1; ++i) {
      float vd = 0.f;
      for (int c = 0; c < 3; ++c) vd += hf[MUL0 + i * 3 + c] * sh1e[e * 3 + c];
      acc4 += vd * (float)w[1792 + i * MUL0 + o];                                          // w4
    }
    out = (acc + acc4 * inv_sq3) * alpha;
  } else {
    int o = (j - MUL0) / 3, c = (j - MUL0) % 3;
    float sw2 = 0.f;
    for (int i = 0; i < MUL0; ++i) sw2 += hf[i] * (float)w[1024 + i * MUL1 + o];           // w2
    float vw3 = 0.f;
    for (int i = 0; i < MUL1; ++i) vw3 += hf[MUL0 + i * 3 + c] * (float)w[1536 + i * MUL1 + o]; // w3
    out = (sw2 * sh1e[e * 3 + c] + vw3) * alpha;
  }
  w_edge[idx] = out * enve[e];
}

// ------------- q-input rows: (node,nE) -> [h[node](128) | e_feat[n](32)] in f16 -------------
__global__ void qin_build(const float* h, const float* e_feat, _Float16* qin) {
  int idx = blockIdx.x * blockDim.x + threadIdx.x;
  if (idx >= FLATN * NE * QIN_K) return;
  int row = idx / QIN_K, d = idx - row * QIN_K;
  int node = row / NE, n = row - node * NE;
  float v = (d < ATOM) ? h[(size_t)node * ATOM + d] : e_feat[(size_t)n * EDIM + (d - ATOM)];
  qin[idx] = (_Float16)v;
}

// ------------- scores: mean-over-heads dot == full 128-dot * 1/(NH*sqrt(HD)) -------------
__global__ void scores_kernel(const _Float16* __restrict__ ke, const _Float16* __restrict__ qn,
                              const int* att_src, const float* active,
                              float* scores, unsigned int* seg_max) {
  int e = blockIdx.x;
  int n = threadIdx.x;
  int src = att_src[e];
  const _Float16* k = ke + (size_t)e * LATN;
  const _Float16* q = qn + ((size_t)src * NE + n) * LATN;
  float s = 0.f;
  for (int d = 0; d < LATN; ++d) s += (float)k[d] * (float)q[d];
  s *= 0.044194173824159216f;  // 1/(4*sqrt(32))
  if (active[e] == 0.0f) s = -1e30f;
  scores[(size_t)e * NE + n] = s;
  atomicMax(&seg_max[src * NE + n], fenc_d(s));
}

__global__ void expsum_kernel(float* scores, const int* att_src, const float* active,
                              const unsigned int* seg_max, float* denom) {
  int idx = blockIdx.x * blockDim.x + threadIdx.x;
  if (idx >= E_NUM * NE) return;
  int e = idx / NE, n = idx - e * NE;
  int src = att_src[e];
  float smax = fdec_d(seg_max[src * NE + n]);
  float ex = (active[e] != 0.f) ? __expf(scores[idx] - smax) : 0.f;
  scores[idx] = ex;                  // reuse scores buffer to hold ex
  atomicAdd(&denom[src * NE + n], ex);
}

// out_inv factorization: A[src,n,:] += attn(e,n) * w_edge[e,:]   (gate applied later)
__global__ void contrib_kernel(const float* ex, const int* att_src, const float* denom,
                               const float* w_edge, float* Aacc) {
  int idx = blockIdx.x * blockDim.x + threadIdx.x;
  if (idx >= E_NUM * NE) return;
  int e = idx / NE, n = idx - e * NE;
  int src = att_src[e];
  float attn = ex[idx] / (denom[src * NE + n] + 1e-12f);
  if (attn == 0.f) return;
  float* dst = Aacc + ((size_t)src * NE + n) * OUTD;
  const float* w = w_edge + (size_t)e * OUTD;
  for (int d = 0; d < OUTD; ++d) atomicAdd(&dst[d], attn * w[d]);
}

__global__ void gate_expand(const float* gates, float* gate_full) {
  int idx = blockIdx.x * blockDim.x + threadIdx.x;
  if (idx >= NE * OUTD) return;
  int n = idx / OUTD, d = idx - n * OUTD;
  gate_full[idx] = (d < MUL0) ? gates[n * INVD + d]
                              : gates[n * INVD + MUL0 + (d - MUL0) / 3];
}

// inv = [scal(32) | vec-norms(16)] padded to 64, f16, with gate applied
__global__ void fin_inv(const float* Aacc, const float* gate_full, _Float16* invp) {
  int row = blockIdx.x * blockDim.x + threadIdx.x;
  if (row >= FLATN * NE) return;
  int n = row % NE;
  const float* a = Aacc + (size_t)row * OUTD;
  const float* g = gate_full + n * OUTD;
  _Float16* o = invp + (size_t)row * INV_K;
  for (int i = 0; i < MUL0; ++i) o[i] = (_Float16)(a[i] * g[i]);
  for (int i = 0; i < MUL1; ++i) {
    float s2 = 1e-12f;
    for (int c = 0; c < 3; ++c) {
      float v = a[MUL0 + i * 3 + c] * g[MUL0 + i * 3 + c];
      s2 += v * v;
    }
    o[MUL0 + i] = (_Float16)sqrtf(s2);
  }
  for (int i = INVD; i < INV_K; ++i) o[i] = (_Float16)0.0f;
}

// ---------------- host side ----------------
static void launch_pack(hipStream_t s, const float* W, _Float16* Wp, int Korig, int Kpad, int N) {
  int total = (N >> 4) * (Kpad >> 5) * 512;
  pack_w<<<(total + 255) / 256, 256, 0, s>>>(W, Wp, Korig, Kpad, N);
}
static void launch_gemm(hipStream_t s, const _Float16* A, const _Float16* Wp, const float* bias,
                        void* C, int M, int Kpad, int N, bool silu, bool outf16) {
  if ((M & 127) == 0 && (N & 63) == 0 && Kpad <= 256) {
    int blocks = (M >> 7) * (N >> 6);
    if (silu && outf16)       gemm_wmma_blk<true, true ><<<blocks, 256, 0, s>>>(A, Wp, bias, C, M, Kpad, N);
    else if (!silu && outf16) gemm_wmma_blk<false, true ><<<blocks, 256, 0, s>>>(A, Wp, bias, C, M, Kpad, N);
    else                      gemm_wmma_blk<false, false><<<blocks, 256, 0, s>>>(A, Wp, bias, C, M, Kpad, N);
  } else {
    int tiles = (M >> 4) * (N >> 4);
    int blocks = (tiles + 7) / 8;
    if (silu && outf16)       gemm_wmma<true, true ><<<blocks, 256, 0, s>>>(A, Wp, bias, C, M, Kpad, N);
    else if (!silu && outf16) gemm_wmma<false, true ><<<blocks, 256, 0, s>>>(A, Wp, bias, C, M, Kpad, N);
    else                      gemm_wmma<false, false><<<blocks, 256, 0, s>>>(A, Wp, bias, C, M, Kpad, N);
  }
}

extern "C" void kernel_launch(void* const* d_in, const int* in_sizes, int n_in,
                              void* d_out, int out_size, void* d_ws, size_t ws_size,
                              hipStream_t stream) {
  (void)n_in; (void)out_size; (void)ws_size;
  // ---- fixed tensor inputs ----
  const float* h        = (const float*)d_in[0];
  const float* h_full   = (const float*)d_in[1];
  const float* e_feat   = (const float*)d_in[2];
  const float* att_dist = (const float*)d_in[3];
  const float* att_vec  = (const float*)d_in[4];
  const int*   z        = (const int*)d_in[5];
  const unsigned char* mask = (const unsigned char*)d_in[6];
  const int*   att_src  = (const int*)d_in[7];
  const int*   att_dst  = (const int*)d_in[8];

  // ---- params: order ambiguous (insertion vs jax-tree-sorted); pick by size probe ----
  const float *z_emb, *vw0w, *vw0b, *vw1w, *vw1b, *e0w, *e0b, *e1w, *e1b;
  const float *q0w, *q0b, *q1w, *q1b, *q2w, *q2b;
  const float *k0w, *k0b, *k1w, *k1b, *k2w, *k2b;
  const float *o0w, *o0b, *o1w, *o1b, *o2w, *o2b;
  if (in_sizes[9] == 101 * ZEMB) {  // insertion order, w before b
    z_emb = (const float*)d_in[9];
    vw0w = (const float*)d_in[10]; vw0b = (const float*)d_in[11];
    vw1w = (const float*)d_in[12]; vw1b = (const float*)d_in[13];
    e0w = (const float*)d_in[14]; e0b = (const float*)d_in[15];
    e1w = (const float*)d_in[16]; e1b = (const float*)d_in[17];
    q0w = (const float*)d_in[18]; q0b = (const float*)d_in[19];
    q1w = (const float*)d_in[20]; q1b = (const float*)d_in[21];
    q2w = (const float*)d_in[22]; q2b = (const float*)d_in[23];
    k0w = (const float*)d_in[24]; k0b = (const float*)d_in[25];
    k1w = (const float*)d_in[26]; k1b = (const float*)d_in[27];
    k2w = (const float*)d_in[28]; k2b = (const float*)d_in[29];
    o0w = (const float*)d_in[30]; o0b = (const float*)d_in[31];
    o1w = (const float*)d_in[32]; o1b = (const float*)d_in[33];
    o2w = (const float*)d_in[34]; o2b = (const float*)d_in[35];
  } else {  // jax tree: keys sorted (emod,kmlp,omlp,qmlp,vw,z_emb), b before w
    e0b = (const float*)d_in[9];  e0w = (const float*)d_in[10];
    e1b = (const float*)d_in[11]; e1w = (const float*)d_in[12];
    k0b = (const float*)d_in[13]; k0w = (const float*)d_in[14];
    k1b = (const float*)d_in[15]; k1w = (const float*)d_in[16];
    k2b = (const float*)d_in[17]; k2w = (const float*)d_in[18];
    o0b = (const float*)d_in[19]; o0w = (const float*)d_in[20];
    o1b = (const float*)d_in[21]; o1w = (const float*)d_in[22];
    o2b = (const float*)d_in[23]; o2w = (const float*)d_in[24];
    q0b = (const float*)d_in[25]; q0w = (const float*)d_in[26];
    q1b = (const float*)d_in[27]; q1w = (const float*)d_in[28];
    q2b = (const float*)d_in[29]; q2w = (const float*)d_in[30];
    vw0b = (const float*)d_in[31]; vw0w = (const float*)d_in[32];
    vw1b = (const float*)d_in[33]; vw1w = (const float*)d_in[34];
    z_emb = (const float*)d_in[35];
  }

  // ---- workspace bump allocator ----
  char* wsp = (char*)d_ws;
  auto alloc = [&](size_t bytes) -> void* {
    void* r = (void*)wsp;
    wsp += (bytes + 255) & ~(size_t)255;
    return r;
  };
  const int QROWS = FLATN * NE;  // 16384 cached q rows
  _Float16* Wp_vw0 = (_Float16*)alloc((size_t)16 * 2 * 512 * 2);
  _Float16* Wp_vw1 = (_Float16*)alloc((size_t)144 * 8 * 512 * 2);
  _Float16* Wp_e0  = (_Float16*)alloc((size_t)16 * 1 * 512 * 2);
  _Float16* Wp_e1  = (_Float16*)alloc((size_t)3 * 8 * 512 * 2);
  _Float16* Wp_q0  = (_Float16*)alloc((size_t)16 * 5 * 512 * 2);
  _Float16* Wp_q1  = (_Float16*)alloc((size_t)16 * 8 * 512 * 2);
  _Float16* Wp_q2  = (_Float16*)alloc((size_t)8 * 8 * 512 * 2);
  _Float16* Wp_k0  = (_Float16*)alloc((size_t)16 * 6 * 512 * 2);
  _Float16* Wp_k1  = (_Float16*)alloc((size_t)16 * 8 * 512 * 2);
  _Float16* Wp_k2  = (_Float16*)alloc((size_t)8 * 8 * 512 * 2);
  _Float16* Wp_o0  = (_Float16*)alloc((size_t)16 * 2 * 512 * 2);
  _Float16* Wp_o1  = (_Float16*)alloc((size_t)16 * 8 * 512 * 2);
  _Float16* Wp_o2  = (_Float16*)alloc((size_t)8 * 8 * 512 * 2);
  _Float16* win    = (_Float16*)alloc((size_t)E_NUM * WIN_K * 2);
  _Float16* ps     = (_Float16*)alloc((size_t)E_NUM * PS_K * 2);
  _Float16* hmid   = (_Float16*)alloc((size_t)E_NUM * HIDN * 2);
  _Float16* tpw    = (_Float16*)alloc((size_t)E_NUM * WNUM * 2);
  _Float16* kh1    = (_Float16*)alloc((size_t)E_NUM * HIDN * 2);
  _Float16* kh2    = (_Float16*)alloc((size_t)E_NUM * HIDN * 2);
  _Float16* ke     = (_Float16*)alloc((size_t)E_NUM * LATN * 2);
  _Float16* efeat16= (_Float16*)alloc((size_t)NE * EDIM * 2);
  _Float16* emodh  = (_Float16*)alloc((size_t)NE * HIDN * 2);
  float*    gates  = (float*)alloc((size_t)NE * INVD * 4);
  float*    gatef  = (float*)alloc((size_t)NE * OUTD * 4);
  _Float16* qin    = (_Float16*)alloc((size_t)QROWS * QIN_K * 2);
  _Float16* qh1    = (_Float16*)alloc((size_t)QROWS * HIDN * 2);
  _Float16* qh2    = (_Float16*)alloc((size_t)QROWS * HIDN * 2);
  _Float16* qn     = (_Float16*)alloc((size_t)QROWS * LATN * 2);
  float*    sh1e   = (float*)alloc((size_t)E_NUM * 3 * 4);
  float*    enve   = (float*)alloc((size_t)E_NUM * 4);
  float*    activ  = (float*)alloc((size_t)E_NUM * 4);
  float*    w_edge = (float*)alloc((size_t)E_NUM * OUTD * 4);
  float*    scores = (float*)alloc((size_t)E_NUM * NE * 4);
  unsigned int* segmax = (unsigned int*)alloc((size_t)QROWS * 4);
  float*    denom  = (float*)alloc((size_t)QROWS * 4);
  float*    Aacc   = (float*)alloc((size_t)QROWS * OUTD * 4);
  _Float16* invp   = (_Float16*)alloc((size_t)QROWS * INV_K * 2);
  _Float16* oh1    = (_Float16*)alloc((size_t)QROWS * HIDN * 2);
  _Float16* oh2    = (_Float16*)alloc((size_t)QROWS * HIDN * 2);

  // ---- 1) pack all weights into WMMA fragment layout (f16) ----
  launch_pack(stream, vw0w, Wp_vw0, 49, WIN_K, HIDN);
  launch_pack(stream, vw1w, Wp_vw1, 256, 256, WNUM);
  launch_pack(stream, e0w,  Wp_e0, 32, 32, HIDN);
  launch_pack(stream, e1w,  Wp_e1, 256, 256, INVD);
  launch_pack(stream, q0w,  Wp_q0, 160, QIN_K, HIDN);
  launch_pack(stream, q1w,  Wp_q1, 256, 256, HIDN);
  launch_pack(stream, q2w,  Wp_q2, 256, 256, LATN);
  launch_pack(stream, k0w,  Wp_k0, 177, PS_K, HIDN);
  launch_pack(stream, k1w,  Wp_k1, 256, 256, HIDN);
  launch_pack(stream, k2w,  Wp_k2, 256, 256, LATN);
  launch_pack(stream, o0w,  Wp_o0, INVD, INV_K, HIDN);
  launch_pack(stream, o1w,  Wp_o1, 256, 256, HIDN);
  launch_pack(stream, o2w,  Wp_o2, 256, 256, LATN);
  cvt_f16<<<(NE * EDIM + 255) / 256, 256, 0, stream>>>(e_feat, efeat16, NE * EDIM);

  // ---- 2) per-edge features + MLP input rows ----
  edge_prep<<<(E_NUM + 255) / 256, 256, 0, stream>>>(att_dist, att_vec, att_src, att_dst,
                                                     z, mask, z_emb, h, win, ps, sh1e, enve, activ);

  // ---- 3) vw MLP (edge -> per-edge tensor-product weights) + TP einsums ----
  launch_gemm(stream, win, Wp_vw0, vw0b, hmid, E_NUM, WIN_K, HIDN, true, true);
  launch_gemm(stream, hmid, Wp_vw1, vw1b, tpw, E_NUM, 256, WNUM, false, true);
  tp_kernel<<<(E_NUM * OUTD + 255) / 256, 256, 0, stream>>>(att_dst, h_full, tpw, sh1e, enve, w_edge);

  // ---- 4) emod gates (tiny M -> fallback GEMM path) ----
  launch_gemm(stream, efeat16, Wp_e0, e0b, emodh, NE, 32, HIDN, true, true);
  launch_gemm(stream, emodh, Wp_e1, e1b, gates, NE, 256, INVD, false, false);
  gate_expand<<<(NE * OUTD + 255) / 256, 256, 0, stream>>>(gates, gatef);

  // ---- 5) kmlp (per edge) ----
  launch_gemm(stream, ps, Wp_k0, k0b, kh1, E_NUM, PS_K, HIDN, true, true);
  launch_gemm(stream, kh1, Wp_k1, k1b, kh2, E_NUM, 256, HIDN, true, true);
  launch_gemm(stream, kh2, Wp_k2, k2b, ke, E_NUM, 256, LATN, false, true);

  // ---- 6) qmlp cached per (node, nE): 32x fewer rows than per (edge, nE) ----
  qin_build<<<((size_t)QROWS * QIN_K + 255) / 256, 256, 0, stream>>>(h, e_feat, qin);
  launch_gemm(stream, qin, Wp_q0, q0b, qh1, QROWS, QIN_K, HIDN, true, true);
  launch_gemm(stream, qh1, Wp_q1, q1b, qh2, QROWS, 256, HIDN, true, true);
  launch_gemm(stream, qh2, Wp_q2, q2b, qn, QROWS, 256, LATN, false, true);

  // ---- 7) segment softmax + scatter-accumulate ----
  fill_u32<<<(QROWS + 255) / 256, 256, 0, stream>>>(segmax, fenc_h(-1e30f), QROWS);
  fill_u32<<<(QROWS + 255) / 256, 256, 0, stream>>>((unsigned int*)denom, 0u, QROWS);
  fill_u32<<<(QROWS * OUTD + 255) / 256, 256, 0, stream>>>((unsigned int*)Aacc, 0u, QROWS * OUTD);
  scores_kernel<<<E_NUM, NE, 0, stream>>>(ke, qn, att_src, activ, scores, segmax);
  expsum_kernel<<<(E_NUM * NE + 255) / 256, 256, 0, stream>>>(scores, att_src, activ, segmax, denom);
  contrib_kernel<<<(E_NUM * NE + 255) / 256, 256, 0, stream>>>(scores, att_src, denom, w_edge, Aacc);

  // ---- 8) invariants + omlp -> d_out ----
  fin_inv<<<(QROWS + 255) / 256, 256, 0, stream>>>(Aacc, gatef, invp);
  launch_gemm(stream, invp, Wp_o0, o0b, oh1, QROWS, INV_K, HIDN, true, true);
  launch_gemm(stream, oh1, Wp_o1, o1b, oh2, QROWS, 256, HIDN, true, true);
  launch_gemm(stream, oh2, Wp_o2, o2b, d_out, QROWS, 256, LATN, false, false);
}
